// Model_29188597743627
// MI455X (gfx1250) — matrix operand order repaired and checked by reading, not compiled, not run on hardware
//
#include <hip/hip_runtime.h>

// ---------------------------------------------------------------------------
// Types for CDNA5 WMMA (gfx1250, wave32)
// ---------------------------------------------------------------------------
typedef __attribute__((ext_vector_type(16))) __bf16 v16bf;
typedef __attribute__((ext_vector_type(8)))  __bf16 v8bf;
typedef __attribute__((ext_vector_type(8)))  float  v8f;

#define NEG_INF (-__builtin_inff())

// Problem constants
#define LSEQ   16384
#define BDIM   768
#define MCAND  2048
#define HDIM   1000
#define HPAD   1024
#define EDIM   2324
#define EPAD   2368      // multiple of 64 (and 32) for WMMA K and N tiling
#define FEATN  20
#define MAXW   30
#define TOPC   50

// ---------------------------------------------------------------------------
// word_attn[l] = bert[l,:] @ W_span + b_span   (one wave32 per row)
// ---------------------------------------------------------------------------
__global__ __launch_bounds__(256) void word_attn_k(
    const float* __restrict__ bert, const float* __restrict__ Wspan,
    const float* __restrict__ bspan, float* __restrict__ wa)
{
    int row  = blockIdx.x * 8 + (threadIdx.x >> 5);
    int lane = threadIdx.x & 31;
    const float* p = bert + (size_t)row * BDIM;
    float acc = 0.f;
    for (int c = lane; c < BDIM; c += 32) acc += p[c] * Wspan[c];
    for (int off = 16; off > 0; off >>= 1) acc += __shfl_down(acc, off, 32);
    if (lane == 0) wa[row] = acc + bspan[0];
}

// ---------------------------------------------------------------------------
// Build mention embeddings (bf16, K padded to EPAD with zeros).
// layout per row: [start(768) | end(768) | width(20) | head(768) | pad(44)]
// ---------------------------------------------------------------------------
__global__ __launch_bounds__(256) void build_ments_k(
    const float* __restrict__ bert, const int* __restrict__ starts,
    const int* __restrict__ widths, const float* __restrict__ width_emb,
    const float* __restrict__ wa, __bf16* __restrict__ X)
{
    int m = blockIdx.x, tid = threadIdx.x;
    int s = starts[m], w = widths[m], e = s + w;
    __shared__ float p[32];
    __shared__ float inv_sh;
    if (tid == 0) {
        float mx = NEG_INF;
        for (int r = 0; r <= w; ++r) mx = fmaxf(mx, wa[s + r]);
        float dn = 0.f;
        for (int r = 0; r <= w; ++r) { float pe = __expf(wa[s + r] - mx); p[r] = pe; dn += pe; }
        inv_sh = 1.f / dn;
    }
    __syncthreads();
    float inv = inv_sh;
    __bf16* xr = X + (size_t)m * EPAD;
    for (int c = tid; c < BDIM; c += 256) {
        xr[c]        = (__bf16)bert[(size_t)s * BDIM + c];
        xr[BDIM + c] = (__bf16)bert[(size_t)e * BDIM + c];
        float acc = 0.f;
        for (int r = 0; r <= w; ++r) acc += p[r] * bert[(size_t)(s + r) * BDIM + c];
        xr[2 * BDIM + FEATN + c] = (__bf16)(acc * inv);
    }
    if (tid < FEATN) xr[2 * BDIM + tid] = (__bf16)width_emb[w * FEATN + tid];
    for (int c = EDIM + tid; c < EPAD; c += 256) xr[c] = (__bf16)0.f;
}

// ---------------------------------------------------------------------------
// width_scores[w] = relu(emb[w]@Ww0 + bw0) @ Ww1 + bw1   (30 blocks)
// ---------------------------------------------------------------------------
__global__ __launch_bounds__(256) void width_scores_k(
    const float* __restrict__ emb, const float* __restrict__ W0,
    const float* __restrict__ b0, const float* __restrict__ W1,
    const float* __restrict__ b1, float* __restrict__ out)
{
    int w = blockIdx.x, tid = threadIdx.x;
    float x[FEATN];
#pragma unroll
    for (int f = 0; f < FEATN; ++f) x[f] = emb[w * FEATN + f];
    float part = 0.f;
    for (int n = tid; n < HDIM; n += 256) {
        float h = b0[n];
#pragma unroll
        for (int f = 0; f < FEATN; ++f) h += x[f] * W0[f * HDIM + n];
        part += fmaxf(h, 0.f) * W1[n];
    }
    __shared__ float red[256];
    red[tid] = part; __syncthreads();
    for (int s = 128; s > 0; s >>= 1) { if (tid < s) red[tid] += red[tid + s]; __syncthreads(); }
    if (tid == 0) out[w] = red[0] + b1[0];
}

// dist_score[b] = dist_emb[b] @ Wd + bd   (10 values)
__global__ void dist_score_k(const float* __restrict__ emb, const float* __restrict__ Wd,
                             const float* __restrict__ bd, float* __restrict__ out)
{
    int b = threadIdx.x;
    if (b < 10) {
        float a = bd[0];
#pragma unroll
        for (int f = 0; f < FEATN; ++f) a += emb[b * FEATN + f] * Wd[f];
        out[b] = a;
    }
}

// ---------------------------------------------------------------------------
// Wt[n][k] = (n<Nreal && k<K) ? W[k*Nreal+n] : 0   (bf16, row-major over K)
// ---------------------------------------------------------------------------
__global__ __launch_bounds__(256) void transpose_bf16_k(
    const float* __restrict__ W, __bf16* __restrict__ Wt,
    int K, int Nreal, int ldt)
{
    int n = blockIdx.y;
    int k = blockIdx.x * 256 + threadIdx.x;
    if (k >= ldt) return;
    float v = (k < K && n < Nreal) ? W[(size_t)k * Nreal + n] : 0.f;
    Wt[(size_t)n * ldt + k] = (__bf16)v;
}

// ---------------------------------------------------------------------------
// WMMA GEMM: C[M,N] = A[M,K] * B[N,K]^T  (B stored row-major over K)
// 256 threads = 8 waves (4x2); block tile 128x64; each wave = 32x32
// (2 A frags x 2 B frags -> 4 accumulators), register double-buffered K loop.
// mode 0: outF = relu(acc + bias[n])            (hidden)
// mode 1: outBF = acc + bias[n]                 (src, bf16)
// mode 2: outF = masked pairwise score          (fast)
// ---------------------------------------------------------------------------
__device__ __forceinline__ int bin_of(int o) {
    // BIN_WIDTHS = [1,1,1,1,1,3,8,16,32,1]
    if (o < 5)  return o;
    if (o < 8)  return 5;
    if (o < 16) return 6;
    if (o < 32) return 7;
    if (o < 64) return 8;
    return 9;
}

union Frag { v16bf v; v8bf h[2]; };

__device__ __forceinline__ void load_frag_a(Frag& f, const __bf16* row, int kk, int aK) {
    f.h[0] = *(const v8bf*)(row + kk + aK);
    f.h[1] = *(const v8bf*)(row + kk + 16 + aK);
}
__device__ __forceinline__ void load_frag_b(Frag& f, const __bf16* row, int kk, int bK) {
    f.h[0] = *(const v8bf*)(row + kk + bK);
    f.h[1] = *(const v8bf*)(row + kk + bK + 16);
}

__global__ __launch_bounds__(256) void gemm_wmma_k(
    const __bf16* __restrict__ A, const __bf16* __restrict__ B,
    int K, int lda, int ldb,
    int mode, float* __restrict__ outF, __bf16* __restrict__ outBF,
    int ldc, int Nreal,
    const float* __restrict__ bias,
    const float* __restrict__ ms,
    const float* __restrict__ distScore)
{
    const int tid  = threadIdx.x;
    const int wave = tid >> 5;
    const int lane = tid & 31;
    const int wm   = wave & 3;        // 0..3 -> 32-row slices
    const int wn   = wave >> 2;       // 0..1 -> 32-col slices
    const int mBase = blockIdx.x * 128 + wm * 32;
    const int nBase = blockIdx.y * 64  + wn * 32;
    const int hlf   = lane >> 4;      // 0/1
    const int l15   = lane & 15;

    v8f acc[2][2];
#pragma unroll
    for (int i = 0; i < 2; ++i)
#pragma unroll
        for (int j = 0; j < 2; ++j) acc[i][j] = (v8f){};

    const __bf16* aR[2] = { A + (size_t)(mBase + l15)      * lda,
                            A + (size_t)(mBase + 16 + l15) * lda };
    const __bf16* bR[2] = { B + (size_t)(nBase + l15)      * ldb,
                            B + (size_t)(nBase + 16 + l15) * ldb };
    const int aK = hlf * 8;   // A frag: elems 0..7 -> K+aK..+7, 8..15 -> K+16+aK..+7
    const int bK = hlf * 8;   // B frag: elems 0..7 -> K+bK..+7, 8..15 -> K+16+bK..+7

    Frag ac[2], bc[2], an[2], bn[2];
#pragma unroll
    for (int i = 0; i < 2; ++i) { load_frag_a(ac[i], aR[i], 0, aK); }
#pragma unroll
    for (int j = 0; j < 2; ++j) { load_frag_b(bc[j], bR[j], 0, bK); }

    for (int kk = 32; kk < K; kk += 32) {
        // issue next tile's loads before consuming current -> overlaps latency
#pragma unroll
        for (int i = 0; i < 2; ++i) { load_frag_a(an[i], aR[i], kk, aK); }
#pragma unroll
        for (int j = 0; j < 2; ++j) { load_frag_b(bn[j], bR[j], kk, bK); }
        __builtin_prefetch(aR[0] + kk + 512, 0, 1);
#pragma unroll
        for (int i = 0; i < 2; ++i)
#pragma unroll
            for (int j = 0; j < 2; ++j)
                acc[i][j] = __builtin_amdgcn_wmma_f32_16x16x32_bf16(
                    false, ac[i].v, false, bc[j].v, (short)0, acc[i][j], false, false);
#pragma unroll
        for (int i = 0; i < 2; ++i) { ac[i] = an[i]; bc[i] = bn[i]; }
    }
#pragma unroll
    for (int i = 0; i < 2; ++i)
#pragma unroll
        for (int j = 0; j < 2; ++j)
            acc[i][j] = __builtin_amdgcn_wmma_f32_16x16x32_bf16(
                false, ac[i].v, false, bc[j].v, (short)0, acc[i][j], false, false);

    // ---- epilogue: 2 row tiles x 2 col tiles; lane holds col (l15), rows +hlf*8
    int col[2] = { nBase + l15, nBase + 16 + l15 };
    int rb[2]  = { mBase + hlf * 8, mBase + 16 + hlf * 8 };

    if (mode == 0) {
        float bv[2];
#pragma unroll
        for (int j = 0; j < 2; ++j) bv[j] = (col[j] < Nreal) ? bias[col[j]] : 0.f;
#pragma unroll
        for (int i = 0; i < 2; ++i)
#pragma unroll
            for (int e = 0; e < 8; ++e) {
                int r = rb[i] + e;
#pragma unroll
                for (int j = 0; j < 2; ++j)
                    outF[(size_t)r * ldc + col[j]] = fmaxf(acc[i][j][e] + bv[j], 0.f);
            }
    } else if (mode == 1) {
        float bv[2];
#pragma unroll
        for (int j = 0; j < 2; ++j) bv[j] = (col[j] < Nreal) ? bias[col[j]] : 0.f;
#pragma unroll
        for (int i = 0; i < 2; ++i)
#pragma unroll
            for (int e = 0; e < 8; ++e) {
                int r = rb[i] + e;
#pragma unroll
                for (int j = 0; j < 2; ++j)
                    outBF[(size_t)r * ldc + col[j]] = (__bf16)(acc[i][j][e] + bv[j]);
            }
    } else {
        float msc[2] = { ms[col[0]], ms[col[1]] };
#pragma unroll
        for (int i = 0; i < 2; ++i)
#pragma unroll
            for (int e = 0; e < 8; ++e) {
                int r = rb[i] + e;
                float msr = ms[r];
#pragma unroll
                for (int j = 0; j < 2; ++j) {
                    int o = r - col[j];
                    float v = (o > 0)
                        ? acc[i][j][e] + msr + msc[j] + distScore[bin_of(o > 64 ? 64 : o)]
                        : NEG_INF;
                    outF[(size_t)r * ldc + col[j]] = v;
                }
            }
    }
}

// ---------------------------------------------------------------------------
// ment_scores[m] = hidden[m,:1000] @ Wm1 + bm1 + width_scores[widths[m]]
// ---------------------------------------------------------------------------
__global__ __launch_bounds__(256) void ment_scores_k(
    const float* __restrict__ hidden, int ldh,
    const float* __restrict__ Wm1, const float* __restrict__ bm1,
    const float* __restrict__ widthScores, const int* __restrict__ widths,
    float* __restrict__ ms)
{
    int m    = blockIdx.x * 8 + (threadIdx.x >> 5);
    int lane = threadIdx.x & 31;
    const float* h = hidden + (size_t)m * ldh;
    float acc = 0.f;
    for (int n = lane; n < HDIM; n += 32) acc += h[n] * Wm1[n];
    for (int off = 16; off > 0; off >>= 1) acc += __shfl_down(acc, off, 32);
    if (lane == 0) ms[m] = acc + bm1[0] + widthScores[widths[m]];
}

// ---------------------------------------------------------------------------
// Top-50 per row, sorted descending (selection via 50 block argmax passes,
// tie-break on lowest index to match lax.top_k).
// ---------------------------------------------------------------------------
__global__ __launch_bounds__(256) void topk_k(
    const float* __restrict__ fast, int ldc, float* __restrict__ out)
{
    int row = blockIdx.x, tid = threadIdx.x;
    __shared__ float vals[MCAND];
    __shared__ float rv[256];
    __shared__ int   ri[256];
    const float* fr = fast + (size_t)row * ldc;
    for (int j = tid; j < MCAND; j += 256) vals[j] = fr[j];
    __syncthreads();
    for (int c = 0; c < TOPC; ++c) {
        float best = NEG_INF; int bi = MCAND;
        for (int j = tid; j < MCAND; j += 256) {
            float v = vals[j];
            if (v > best || (v == best && j < bi)) { best = v; bi = j; }
        }
        rv[tid] = best; ri[tid] = bi; __syncthreads();
        for (int s = 128; s > 0; s >>= 1) {
            if (tid < s) {
                if (rv[tid + s] > rv[tid] ||
                    (rv[tid + s] == rv[tid] && ri[tid + s] < ri[tid])) {
                    rv[tid] = rv[tid + s]; ri[tid] = ri[tid + s];
                }
            }
            __syncthreads();
        }
        if (tid == 0) {
            out[(size_t)row * TOPC + c] = rv[0];
            if (ri[0] < MCAND) vals[ri[0]] = NEG_INF;
        }
        __syncthreads();
    }
}

// ---------------------------------------------------------------------------
// Host launcher
// ---------------------------------------------------------------------------
extern "C" void kernel_launch(void* const* d_in, const int* in_sizes, int n_in,
                              void* d_out, int out_size, void* d_ws, size_t ws_size,
                              hipStream_t stream)
{
    (void)in_sizes; (void)n_in; (void)out_size; (void)ws_size;
    const float* bert    = (const float*)d_in[0];
    const int*   starts  = (const int*)  d_in[1];
    const int*   widths  = (const int*)  d_in[2];
    const float* Wspan   = (const float*)d_in[3];
    const float* bspan   = (const float*)d_in[4];
    const float* wemb    = (const float*)d_in[5];
    const float* Wm0     = (const float*)d_in[6];
    const float* bm0     = (const float*)d_in[7];
    const float* Wm1     = (const float*)d_in[8];
    const float* bm1     = (const float*)d_in[9];
    const float* Ww0     = (const float*)d_in[10];
    const float* bw0     = (const float*)d_in[11];
    const float* Ww1     = (const float*)d_in[12];
    const float* bw1     = (const float*)d_in[13];
    const float* wse     = (const float*)d_in[14];
    const float* W_fast  = (const float*)d_in[15];
    const float* b_fast  = (const float*)d_in[16];
    const float* Wd      = (const float*)d_in[17];
    const float* bd      = (const float*)d_in[18];
    const float* dse     = (const float*)d_in[19];

    char* base = (char*)d_ws;
    size_t off = 0;
    auto alloc = [&](size_t bytes) -> char* {
        char* p = base + off;
        off = (off + bytes + 255) & ~(size_t)255;
        return p;
    };
    float*  wa          = (float*) alloc((size_t)LSEQ * 4);
    float*  widthScores = (float*) alloc(32 * 4);
    float*  distScore   = (float*) alloc(16 * 4);
    float*  ms          = (float*) alloc((size_t)MCAND * 4);
    __bf16* Xbf         = (__bf16*)alloc((size_t)MCAND * EPAD * 2);
    __bf16* Wm0t        = (__bf16*)alloc((size_t)HPAD  * EPAD * 2);
    __bf16* Wft         = (__bf16*)alloc((size_t)EPAD  * EPAD * 2);
    __bf16* srcBf       = (__bf16*)alloc((size_t)MCAND * EPAD * 2);
    float*  hidden      = (float*) alloc((size_t)MCAND * HPAD * 4);
    float*  fast        = (float*) alloc((size_t)MCAND * MCAND * 4);

    // stage 0: word attention logits
    word_attn_k<<<LSEQ / 8, 256, 0, stream>>>(bert, Wspan, bspan, wa);
    // stage 1: mention embeddings (bf16, padded)
    build_ments_k<<<MCAND, 256, 0, stream>>>(bert, starts, widths, wemb, wa, Xbf);
    // tiny scorers
    width_scores_k<<<MAXW, 256, 0, stream>>>(wse, Ww0, bw0, Ww1, bw1, widthScores);
    dist_score_k<<<1, 32, 0, stream>>>(dse, Wd, bd, distScore);
    // weight transposes (row-major over K, bf16, zero padded)
    transpose_bf16_k<<<dim3((EPAD + 255) / 256, HPAD), 256, 0, stream>>>(Wm0, Wm0t, EDIM, HDIM, EPAD);
    transpose_bf16_k<<<dim3((EPAD + 255) / 256, EPAD), 256, 0, stream>>>(W_fast, Wft, EDIM, EDIM, EPAD);
    // GEMM1: hidden = relu(X @ Wm0 + bm0)
    gemm_wmma_k<<<dim3(MCAND / 128, HPAD / 64), 256, 0, stream>>>(
        Xbf, Wm0t, EPAD, EPAD, EPAD, 0, hidden, nullptr, HPAD, HDIM,
        bm0, nullptr, nullptr);
    // mention scores
    ment_scores_k<<<MCAND / 8, 256, 0, stream>>>(hidden, HPAD, Wm1, bm1, widthScores, widths, ms);
    // GEMM2: src = X @ W_fast + b_fast  (bf16 out)
    gemm_wmma_k<<<dim3(MCAND / 128, EPAD / 64), 256, 0, stream>>>(
        Xbf, Wft, EPAD, EPAD, EPAD, 1, nullptr, srcBf, EPAD, EDIM,
        b_fast, nullptr, nullptr);
    // GEMM3: fast = src @ X^T + scores + dist, causal masked
    gemm_wmma_k<<<dim3(MCAND / 128, MCAND / 64), 256, 0, stream>>>(
        srcBf, Xbf, EPAD, EPAD, EPAD, 2, fast, nullptr, MCAND, MCAND,
        nullptr, ms, distScore);
    // top-50 per row
    topk_k<<<MCAND, 256, 0, stream>>>(fast, MCAND, (float*)d_out);
}
